// ComplexEqModel_38027640438969
// MI455X (gfx1250) — matrix operand order repaired and checked by reading, not compiled, not run on hardware
//
#include <hip/hip_runtime.h>

// ---------------- problem constants ----------------
constexpr int cB = 4, cT = 1024, cNF = 7, cFD = 128;
constexpr int cD = 896, cH = 8, cHD = 112, cNL = 6;
constexpr int cVT = 1001, cVO = 1001, cFF = 3584;
constexpr int cM = cB * cT;                       // 4096 tokens
constexpr float cEPS = 1e-5f;
constexpr float cSCALE = 0.09449111825230680775f; // 112^-0.5

// ---------------- types ----------------
typedef __bf16 bf16;
typedef __attribute__((ext_vector_type(16))) __bf16 v16bf;
typedef __attribute__((ext_vector_type(8)))  __bf16 v8bf;
typedef __attribute__((ext_vector_type(8)))  float  v8f;
typedef __attribute__((ext_vector_type(16))) unsigned short v16us;

__device__ inline v8f zero8() {
  v8f z = {0.f, 0.f, 0.f, 0.f, 0.f, 0.f, 0.f, 0.f};
  return z;
}

__device__ inline v8f wmma_bf16(v16bf a, v16bf b, v8f c) {
  // D = A(16x32) * B(32x16) + C, fp32 accumulate
  return __builtin_amdgcn_wmma_f32_16x16x32_bf16(false, a, false, b, (short)0, c,
                                                 false, false);
}

// packed bf16 negation (f16/bf16 WMMA has no A/B NEG per ISA -> flip sign bits)
__device__ inline v16bf negb(v16bf x) {
  v16us u = __builtin_bit_cast(v16us, x);
  u ^= (unsigned short)0x8000;
  return __builtin_bit_cast(v16bf, u);
}

// A fragment (16x32, row-major src, leading dim ld): lane m = lane&15,
// halves 0..7 -> K kb..kb+7, halves 8..15 -> K kb+16..kb+23, kb = k0 + (lane>=16 ? 8 : 0)
__device__ inline v16bf load_a_frag(const bf16* __restrict__ base, int ld, int m0, int k0) {
  int lane = threadIdx.x & 31;
  const bf16* p = base + (size_t)(m0 + (lane & 15)) * ld + k0 + ((lane & 16) ? 8 : 0);
  v8bf lo = *(const v8bf*)p;
  v8bf hi = *(const v8bf*)(p + 16);
  v16bf f;
#pragma unroll
  for (int i = 0; i < 8; ++i) { f[i] = lo[i]; f[i + 8] = hi[i]; }
  return f;
}

// B fragment (32x16): lane n = n0 + (lane&15); halves 0..7 -> K kb..kb+7, halves 8..15 ->
// K kb+16..kb+23, kb = k0 + (lane>=16 ? 8 : 0).  src row-major [N,K] (i.e. B = src^T, matches x @ W.T)
__device__ inline v16bf load_b_frag(const bf16* __restrict__ base, int ld, int n0, int k0,
                                    int nmax) {
  int lane = threadIdx.x & 31;
  int n = n0 + (lane & 15);
  if (n > nmax) n = nmax;
  const bf16* p = base + (size_t)n * ld + k0 + ((lane & 16) ? 8 : 0);
  v8bf lo = *(const v8bf*)p;
  v8bf hi = *(const v8bf*)(p + 16);
  v16bf f;
#pragma unroll
  for (int i = 0; i < 8; ++i) { f[i] = lo[i]; f[i + 8] = hi[i]; }
  return f;
}

__device__ inline void wait_ds0() {
#if __has_builtin(__builtin_amdgcn_s_wait_dscnt)
  __builtin_amdgcn_s_wait_dscnt(0);
#else
  asm volatile("s_wait_dscnt 0" ::: "memory");
#endif
}

// ---------------- elementwise kernels ----------------
__global__ void cvt_kernel(const float* __restrict__ src, bf16* __restrict__ dst, size_t n) {
  size_t i = (size_t)blockIdx.x * blockDim.x + threadIdx.x;
  size_t stride = (size_t)gridDim.x * blockDim.x;
  for (; i < n; i += stride) dst[i] = (bf16)src[i];
}

__global__ void embed_kernel(const int* __restrict__ data,
                             const float* __restrict__ emb_r, const float* __restrict__ emb_i,
                             const float* __restrict__ pe_r, const float* __restrict__ pe_i,
                             float* __restrict__ xr, float* __restrict__ xi) {
  size_t idx = (size_t)blockIdx.x * blockDim.x + threadIdx.x;
  if (idx >= (size_t)cM * cD) return;
  int c = (int)(idx % cD);
  size_t bt = idx / cD;
  int t = (int)(bt & (cT - 1));
  int f = c / cFD, d = c % cFD;
  int tok = data[bt * cNF + f];
  size_t e = ((size_t)f * cVT + tok) * cFD + d;
  size_t pe = (size_t)t * cD + c;
  xr[idx] = emb_r[e] + pe_r[pe];
  xi[idx] = emb_i[e] + pe_i[pe];
}

// complex LayerNorm: one wave32 per token, 28 channels per lane
__global__ __launch_bounds__(256) void c_layernorm_kernel(
    const float* __restrict__ xr, const float* __restrict__ xi,
    const float* __restrict__ gr, const float* __restrict__ gi,
    const float* __restrict__ br, const float* __restrict__ bi,
    bf16* __restrict__ outR, bf16* __restrict__ outI) {
  int tok = blockIdx.x * 8 + (threadIdx.x >> 5);
  int lane = threadIdx.x & 31;
  const float* pr = xr + (size_t)tok * cD;
  const float* pi = xi + (size_t)tok * cD;
  float vr[28], vi[28];
  float sr = 0.f, si = 0.f;
#pragma unroll
  for (int j = 0; j < 28; ++j) {
    int c = lane + j * 32;
    vr[j] = pr[c]; vi[j] = pi[c];
    sr += vr[j]; si += vi[j];
  }
#pragma unroll
  for (int o = 16; o > 0; o >>= 1) { sr += __shfl_xor(sr, o, 32); si += __shfl_xor(si, o, 32); }
  float mr = sr * (1.f / cD), mi_ = si * (1.f / cD);
  float sv = 0.f;
#pragma unroll
  for (int j = 0; j < 28; ++j) {
    vr[j] -= mr; vi[j] -= mi_;
    sv += vr[j] * vr[j] + vi[j] * vi[j];
  }
#pragma unroll
  for (int o = 16; o > 0; o >>= 1) sv += __shfl_xor(sv, o, 32);
  float inv = rsqrtf(sv * (1.f / cD) + cEPS);
  bf16* oR = outR + (size_t)tok * cD;
  bf16* oI = outI + (size_t)tok * cD;
#pragma unroll
  for (int j = 0; j < 28; ++j) {
    int c = lane + j * 32;
    float rn = vr[j] * inv, im = vi[j] * inv;
    float g_r = gr[c], g_i = gi[c];
    oR[c] = (bf16)(g_r * rn - g_i * im + br[c]);
    oI[c] = (bf16)(g_r * im + g_i * rn + bi[c]);
  }
}

// ---------------- complex GEMM ----------------
// C = (Ar + i Ai) @ (Wr + i Wi)^T   with A:[M,K] bf16, W:[N,K] bf16
// block: 256 thr = 8 waves as 2(M) x 4(N); wave tile 32x32 (2x2 WMMA subtiles)
// Software-pipelined: next K-slab's fragments are loaded before this slab's WMMAs.
// MODE 0: bf16 out [M,N] (+bias, opt ReLU)        (FF1)
// MODE 1: fp32 residual accumulate out += (+bias) (attn-proj, FF2)
// MODE 2: fp32 interleaved (re,im) head output, guard n<N (head)
// MODE 3: Q/K scatter to [B,H,T,128] padded bf16
// MODE 4: V scatter transposed to [B,H,HD,T] bf16
template <int MODE, bool RELU>
__global__ __launch_bounds__(256) void cgemm_kernel(
    const bf16* __restrict__ Ar, const bf16* __restrict__ Ai,
    const bf16* __restrict__ Wr, const bf16* __restrict__ Wi,
    const float* __restrict__ bias_r, const float* __restrict__ bias_i,
    float* __restrict__ outRf, float* __restrict__ outIf,
    bf16* __restrict__ outRb, bf16* __restrict__ outIb,
    int M, int N, int K) {
  int wave = threadIdx.x >> 5;
  int lane = threadIdx.x & 31;
  int m0 = blockIdx.x * 64 + (wave & 1) * 32;
  int n0 = blockIdx.y * 128 + (wave >> 1) * 32;
  int nmax = N - 1;

  v8f aR[2][2], aI[2][2];
#pragma unroll
  for (int i = 0; i < 2; ++i)
#pragma unroll
    for (int j = 0; j < 2; ++j) { aR[i][j] = zero8(); aI[i][j] = zero8(); }

  // prologue: fragments for k0 = 0
  v16bf arC[2], aiC[2], wrC[2], wiC[2];
#pragma unroll
  for (int mi = 0; mi < 2; ++mi) {
    arC[mi] = load_a_frag(Ar, K, m0 + mi * 16, 0);
    aiC[mi] = load_a_frag(Ai, K, m0 + mi * 16, 0);
  }
#pragma unroll
  for (int ni = 0; ni < 2; ++ni) {
    wrC[ni] = load_b_frag(Wr, K, n0 + ni * 16, 0, nmax);
    wiC[ni] = load_b_frag(Wi, K, n0 + ni * 16, 0, nmax);
  }

  for (int k0 = 0; k0 < K; k0 += 32) {
    int kn = k0 + 32;
    bool more = kn < K;
    // issue next-slab loads BEFORE this slab's WMMAs (overlap latency)
    v16bf arN[2], aiN[2], wrN[2], wiN[2];
    if (more) {
#pragma unroll
      for (int mi = 0; mi < 2; ++mi) {
        arN[mi] = load_a_frag(Ar, K, m0 + mi * 16, kn);
        aiN[mi] = load_a_frag(Ai, K, m0 + mi * 16, kn);
      }
#pragma unroll
      for (int ni = 0; ni < 2; ++ni) {
        wrN[ni] = load_b_frag(Wr, K, n0 + ni * 16, kn, nmax);
        wiN[ni] = load_b_frag(Wi, K, n0 + ni * 16, kn, nmax);
      }
      if (kn + 32 < K) {  // L2 prefetch two slabs ahead
        int npf = n0 + (lane & 15);
        if (npf > nmax) npf = nmax;
        __builtin_prefetch(Wr + (size_t)npf * K + kn + 32, 0, 1);
      }
    }
    v16bf anC[2];
#pragma unroll
    for (int mi = 0; mi < 2; ++mi) anC[mi] = negb(aiC[mi]);
#pragma unroll
    for (int ni = 0; ni < 2; ++ni)
#pragma unroll
      for (int mi = 0; mi < 2; ++mi) {
        aR[mi][ni] = wmma_bf16(arC[mi], wrC[ni], aR[mi][ni]);
        aR[mi][ni] = wmma_bf16(anC[mi], wiC[ni], aR[mi][ni]);  // - Ai*Wi
        aI[mi][ni] = wmma_bf16(arC[mi], wiC[ni], aI[mi][ni]);
        aI[mi][ni] = wmma_bf16(aiC[mi], wrC[ni], aI[mi][ni]);
      }
    if (more) {
#pragma unroll
      for (int mi = 0; mi < 2; ++mi) { arC[mi] = arN[mi]; aiC[mi] = aiN[mi]; }
#pragma unroll
      for (int ni = 0; ni < 2; ++ni) { wrC[ni] = wrN[ni]; wiC[ni] = wiN[ni]; }
    }
  }

#pragma unroll
  for (int mi = 0; mi < 2; ++mi)
#pragma unroll
    for (int ni = 0; ni < 2; ++ni) {
      int n = n0 + ni * 16 + (lane & 15);
      int mb = m0 + mi * 16 + ((lane & 16) ? 8 : 0);
      int nc = n > nmax ? nmax : n;
      float fbr = bias_r ? bias_r[nc] : 0.f;
      float fbi = bias_i ? bias_i[nc] : 0.f;
#pragma unroll
      for (int r = 0; r < 8; ++r) {
        int m = mb + r;
        float cr = aR[mi][ni][r] + fbr;
        float ci = aI[mi][ni][r] + fbi;
        if (RELU) { cr = fmaxf(cr, 0.f); ci = fmaxf(ci, 0.f); }
        if constexpr (MODE == 0) {
          size_t o = (size_t)m * N + n;
          outRb[o] = (bf16)cr; outIb[o] = (bf16)ci;
        } else if constexpr (MODE == 1) {
          size_t o = (size_t)m * N + n;
          outRf[o] += cr; outIf[o] += ci;
        } else if constexpr (MODE == 2) {
          if (n < N) {
            size_t o = ((size_t)m * N + n) * 2;
            outRf[o] = cr; outRf[o + 1] = ci;
          }
        } else if constexpr (MODE == 3) {
          int b_ = m >> 10, t = m & (cT - 1);
          int h_ = n / cHD, d = n % cHD;
          size_t o = (((size_t)(b_ * cH + h_)) * cT + t) * 128 + d;
          outRb[o] = (bf16)cr; outIb[o] = (bf16)ci;
        } else if constexpr (MODE == 4) {
          int b_ = m >> 10, t = m & (cT - 1);
          int h_ = n / cHD, d = n % cHD;
          size_t o = (((size_t)(b_ * cH + h_)) * cHD + d) * cT + t;
          outRb[o] = (bf16)cr; outIb[o] = (bf16)ci;
        }
      }
    }
}

// ---------------- flash-style causal complex attention ----------------
// Q,K: [B,H,T,128] bf16 (HD=112 zero-padded to 128); V: [B,H,HD,T] bf16 (transposed)
// one wave per 16-query tile; S = QrKr^T + QiKi^T via WMMA; online softmax;
// P goes through a wave-private LDS tile to convert C-layout -> A-layout.
// Inner loops software-pipelined like the GEMM.
__global__ __launch_bounds__(256) void attention_kernel(
    const bf16* __restrict__ q_r, const bf16* __restrict__ q_i,
    const bf16* __restrict__ k_r, const bf16* __restrict__ k_i,
    const bf16* __restrict__ vt_r, const bf16* __restrict__ vt_i,
    bf16* __restrict__ ao_r, bf16* __restrict__ ao_i) {
  __shared__ __align__(16) bf16 lds_p[8][16 * 32];
  int wave = threadIdx.x >> 5;
  int lane = threadIdx.x & 31;
  int q0 = (blockIdx.x * 8 + wave) * 16;
  int bh = blockIdx.y;
  int b_ = bh >> 3, h_ = bh & 7;
  const bf16* qr_b = q_r + (size_t)bh * cT * 128;
  const bf16* qi_b = q_i + (size_t)bh * cT * 128;
  const bf16* kr_b = k_r + (size_t)bh * cT * 128;
  const bf16* ki_b = k_i + (size_t)bh * cT * 128;
  const bf16* vr_b = vt_r + (size_t)bh * cHD * cT;
  const bf16* vi_b = vt_i + (size_t)bh * cHD * cT;

  v16bf fqr[4], fqi[4];
#pragma unroll
  for (int kk = 0; kk < 4; ++kk) {
    fqr[kk] = load_a_frag(qr_b, 128, q0, kk * 32);
    fqi[kk] = load_a_frag(qi_b, 128, q0, kk * 32);
  }

  float mrun[8], lrun[8];
  v8f Or[7], Oi[7];
#pragma unroll
  for (int r = 0; r < 8; ++r) { mrun[r] = -1e30f; lrun[r] = 0.f; }
#pragma unroll
  for (int j = 0; j < 7; ++j) { Or[j] = zero8(); Oi[j] = zero8(); }

  int n_ = lane & 15;
  int moff = (lane & 16) ? 8 : 0;
  bf16* lp = lds_p[wave];

  for (int k0 = 0; k0 < q0 + 16; k0 += 32) {
    v8f s0 = zero8(), s1 = zero8();
    // ---- S = Qr Kr^T + Qi Ki^T (pipelined over the 4 HD slabs) ----
    v16bf c0r = load_b_frag(kr_b, 128, k0, 0, cT - 1);
    v16bf c0i = load_b_frag(ki_b, 128, k0, 0, cT - 1);
    v16bf c1r = load_b_frag(kr_b, 128, k0 + 16, 0, cT - 1);
    v16bf c1i = load_b_frag(ki_b, 128, k0 + 16, 0, cT - 1);
#pragma unroll
    for (int kk = 0; kk < 4; ++kk) {
      v16bf d0r, d0i, d1r, d1i;
      if (kk < 3) {
        d0r = load_b_frag(kr_b, 128, k0, (kk + 1) * 32, cT - 1);
        d0i = load_b_frag(ki_b, 128, k0, (kk + 1) * 32, cT - 1);
        d1r = load_b_frag(kr_b, 128, k0 + 16, (kk + 1) * 32, cT - 1);
        d1i = load_b_frag(ki_b, 128, k0 + 16, (kk + 1) * 32, cT - 1);
      }
      s0 = wmma_bf16(fqr[kk], c0r, s0);
      s0 = wmma_bf16(fqi[kk], c0i, s0);
      s1 = wmma_bf16(fqr[kk], c1r, s1);
      s1 = wmma_bf16(fqi[kk], c1i, s1);
      if (kk < 3) { c0r = d0r; c0i = d0i; c1r = d1r; c1i = d1i; }
    }
    bool need_mask = (k0 + 31 > q0);
#pragma unroll
    for (int r = 0; r < 8; ++r) {
      float a0 = s0[r] * cSCALE, a1 = s1[r] * cSCALE;
      if (need_mask) {
        int qrow = q0 + moff + r;
        if (k0 + n_ > qrow) a0 = -1e30f;
        if (k0 + 16 + n_ > qrow) a1 = -1e30f;
      }
      float rm = fmaxf(a0, a1);
#pragma unroll
      for (int o = 8; o > 0; o >>= 1) rm = fmaxf(rm, __shfl_xor(rm, o, 32));
      float nm = fmaxf(mrun[r], rm);
      float alpha = __expf(mrun[r] - nm);
      float p0 = __expf(a0 - nm), p1 = __expf(a1 - nm);
      float rs = p0 + p1;
#pragma unroll
      for (int o = 8; o > 0; o >>= 1) rs += __shfl_xor(rs, o, 32);
      lrun[r] = lrun[r] * alpha + rs;
      mrun[r] = nm;
#pragma unroll
      for (int j = 0; j < 7; ++j) { Or[j][r] *= alpha; Oi[j][r] *= alpha; }
      int row = moff + r;
      lp[row * 32 + n_] = (bf16)p0;
      lp[row * 32 + 16 + n_] = (bf16)p1;
    }
    wait_ds0();
    // reload P tile in A-fragment layout (wave-private region, DS pipe is in-order)
    const bf16* pp = lp + (lane & 15) * 32 + ((lane & 16) ? 8 : 0);
    v8bf plo = *(const v8bf*)pp;
    v8bf phi = *(const v8bf*)(pp + 16);
    v16bf fp;
#pragma unroll
    for (int i = 0; i < 8; ++i) { fp[i] = plo[i]; fp[i + 8] = phi[i]; }
    // ---- O += P V (pipelined over the 7 HD tiles) ----
    v16bf cvr = load_b_frag(vr_b, cT, 0, k0, cHD - 1);
    v16bf cvi = load_b_frag(vi_b, cT, 0, k0, cHD - 1);
#pragma unroll
    for (int j = 0; j < 7; ++j) {
      v16bf dvr, dvi;
      if (j < 6) {
        dvr = load_b_frag(vr_b, cT, (j + 1) * 16, k0, cHD - 1);
        dvi = load_b_frag(vi_b, cT, (j + 1) * 16, k0, cHD - 1);
      }
      Or[j] = wmma_bf16(fp, cvr, Or[j]);
      Oi[j] = wmma_bf16(fp, cvi, Oi[j]);
      if (j < 6) { cvr = dvr; cvi = dvi; }
    }
  }

#pragma unroll
  for (int r = 0; r < 8; ++r) {
    float invl = 1.f / lrun[r];
    int t = q0 + moff + r;
    size_t rowbase = ((size_t)b_ * cT + t) * cD + (size_t)h_ * cHD;
#pragma unroll
    for (int j = 0; j < 7; ++j) {
      ao_r[rowbase + j * 16 + n_] = (bf16)(Or[j][r] * invl);
      ao_i[rowbase + j * 16 + n_] = (bf16)(Oi[j][r] * invl);
    }
  }
}

// ---------------- workspace layout ----------------
constexpr size_t ALGN(size_t x) { return (x + 255) & ~(size_t)255; }
constexpr size_t SZ_DD = ALGN((size_t)cD * cD * 2);         // one D*D bf16 matrix
constexpr size_t SZ_FD = ALGN((size_t)cFF * cD * 2);        // one FF*D bf16 matrix
constexpr size_t SZ_WH = ALGN((size_t)cVO * cD * 2);
constexpr size_t SZ_XF = ALGN((size_t)cM * cD * 4);         // fp32 residual
constexpr size_t SZ_XB = ALGN((size_t)cM * cD * 2);         // bf16 activation
constexpr size_t SZ_QK = ALGN((size_t)cB * cH * cT * 128 * 2);
constexpr size_t SZ_VT = ALGN((size_t)cB * cH * cHD * cT * 2);
constexpr size_t SZ_HH = ALGN((size_t)cM * cFF * 2);
constexpr size_t WS_NEEDED = 8 * SZ_DD + 4 * SZ_FD + 2 * SZ_WH + 2 * SZ_XF +
                             2 * SZ_XB + 4 * SZ_QK + 2 * SZ_VT + 2 * SZ_XB + 2 * SZ_HH;

extern "C" void kernel_launch(void* const* d_in, const int* in_sizes, int n_in,
                              void* d_out, int out_size, void* d_ws, size_t ws_size,
                              hipStream_t stream) {
  (void)in_sizes; (void)n_in; (void)out_size;
  if (ws_size < WS_NEEDED) return;

  // ---- inputs (setup_inputs dict order) ----
  const int*   data   = (const int*)d_in[0];
  const float* emb_r  = (const float*)d_in[1];
  const float* emb_i  = (const float*)d_in[2];
  const float* pe_r   = (const float*)d_in[3];
  const float* pe_i   = (const float*)d_in[4];
  const float* ln1_gr = (const float*)d_in[5];
  const float* ln1_gi = (const float*)d_in[6];
  const float* ln1_br = (const float*)d_in[7];
  const float* ln1_bi = (const float*)d_in[8];
  const float* wq_r   = (const float*)d_in[9];
  const float* wq_i   = (const float*)d_in[10];
  const float* wk_r   = (const float*)d_in[11];
  const float* wk_i   = (const float*)d_in[12];
  const float* wv_r   = (const float*)d_in[13];
  const float* wv_i   = (const float*)d_in[14];
  const float* wo_r   = (const float*)d_in[15];
  const float* wo_i   = (const float*)d_in[16];
  const float* bo_r   = (const float*)d_in[17];
  const float* bo_i   = (const float*)d_in[18];
  const float* ln2_gr = (const float*)d_in[19];
  const float* ln2_gi = (const float*)d_in[20];
  const float* ln2_br = (const float*)d_in[21];
  const float* ln2_bi = (const float*)d_in[22];
  const float* w1_r   = (const float*)d_in[23];
  const float* w1_i   = (const float*)d_in[24];
  const float* b1_r   = (const float*)d_in[25];
  const float* b1_i   = (const float*)d_in[26];
  const float* w2_r   = (const float*)d_in[27];
  const float* w2_i   = (const float*)d_in[28];
  const float* b2_r   = (const float*)d_in[29];
  const float* b2_i   = (const float*)d_in[30];
  const float* lnf_gr = (const float*)d_in[31];
  const float* lnf_gi = (const float*)d_in[32];
  const float* lnf_br = (const float*)d_in[33];
  const float* lnf_bi = (const float*)d_in[34];
  const float* wh_r   = (const float*)d_in[35];
  const float* wh_i   = (const float*)d_in[36];
  const float* bh_r   = (const float*)d_in[37];
  const float* bh_i   = (const float*)d_in[38];

  // ---- workspace carve-up ----
  char* ws = (char*)d_ws;
  auto take = [&](size_t bytes) { char* p = ws; ws += bytes; return p; };
  bf16* s_wq_r = (bf16*)take(SZ_DD); bf16* s_wq_i = (bf16*)take(SZ_DD);
  bf16* s_wk_r = (bf16*)take(SZ_DD); bf16* s_wk_i = (bf16*)take(SZ_DD);
  bf16* s_wv_r = (bf16*)take(SZ_DD); bf16* s_wv_i = (bf16*)take(SZ_DD);
  bf16* s_wo_r = (bf16*)take(SZ_DD); bf16* s_wo_i = (bf16*)take(SZ_DD);
  bf16* s_w1_r = (bf16*)take(SZ_FD); bf16* s_w1_i = (bf16*)take(SZ_FD);
  bf16* s_w2_r = (bf16*)take(SZ_FD); bf16* s_w2_i = (bf16*)take(SZ_FD);
  bf16* s_wh_r = (bf16*)take(SZ_WH); bf16* s_wh_i = (bf16*)take(SZ_WH);
  float* xr = (float*)take(SZ_XF);  float* xi = (float*)take(SZ_XF);
  bf16* nr = (bf16*)take(SZ_XB);    bf16* ni = (bf16*)take(SZ_XB);
  bf16* qr = (bf16*)take(SZ_QK);    bf16* qi = (bf16*)take(SZ_QK);
  bf16* kr = (bf16*)take(SZ_QK);    bf16* ki = (bf16*)take(SZ_QK);
  bf16* vtr = (bf16*)take(SZ_VT);   bf16* vti = (bf16*)take(SZ_VT);
  bf16* aor = (bf16*)take(SZ_XB);   bf16* aoi = (bf16*)take(SZ_XB);
  bf16* hr = (bf16*)take(SZ_HH);    bf16* hi = (bf16*)take(SZ_HH);

  auto cvt = [&](const float* s, bf16* d, size_t n) {
    int bl = (int)((n + 2047) / 2048);
    if (bl > 4096) bl = 4096;
    cvt_kernel<<<bl, 256, 0, stream>>>(s, d, n);
  };

  dim3 blk(256);
  dim3 g_d(cM / 64, (cD + 127) / 128);
  dim3 g_ff(cM / 64, (cFF + 127) / 128);
  dim3 g_vo(cM / 64, (cVO + 127) / 128);
  dim3 g_att(cT / 128, cB * cH);
  int g_ln = cM / 8;

  // zero the HD->128 pad lanes of Q/K once per launch
  size_t qkBytes = (size_t)cB * cH * cT * 128 * 2;
  hipMemsetAsync(qr, 0, qkBytes, stream);
  hipMemsetAsync(qi, 0, qkBytes, stream);
  hipMemsetAsync(kr, 0, qkBytes, stream);
  hipMemsetAsync(ki, 0, qkBytes, stream);

  // embeddings + positional encoding -> fp32 residual stream
  {
    size_t n = (size_t)cM * cD;
    embed_kernel<<<(int)((n + 255) / 256), 256, 0, stream>>>(data, emb_r, emb_i, pe_r, pe_i,
                                                             xr, xi);
  }

  // head weights (converted once)
  cvt(wh_r, s_wh_r, (size_t)cVO * cD);
  cvt(wh_i, s_wh_i, (size_t)cVO * cD);

  const size_t DD = (size_t)cD * cD;
  const size_t FD_ = (size_t)cFF * cD;

  for (int l = 0; l < cNL; ++l) {
    // stage this layer's weights as bf16 (stream-ordered reuse of the staging buffers)
    cvt(wq_r + l * DD, s_wq_r, DD); cvt(wq_i + l * DD, s_wq_i, DD);
    cvt(wk_r + l * DD, s_wk_r, DD); cvt(wk_i + l * DD, s_wk_i, DD);
    cvt(wv_r + l * DD, s_wv_r, DD); cvt(wv_i + l * DD, s_wv_i, DD);
    cvt(wo_r + l * DD, s_wo_r, DD); cvt(wo_i + l * DD, s_wo_i, DD);
    cvt(w1_r + l * FD_, s_w1_r, FD_); cvt(w1_i + l * FD_, s_w1_i, FD_);
    cvt(w2_r + l * FD_, s_w2_r, FD_); cvt(w2_i + l * FD_, s_w2_i, FD_);

    // LN1
    c_layernorm_kernel<<<g_ln, blk, 0, stream>>>(xr, xi,
        ln1_gr + l * cD, ln1_gi + l * cD, ln1_br + l * cD, ln1_bi + l * cD, nr, ni);

    // Q, K (MODE3 -> padded [B,H,T,128]), V (MODE4 -> transposed [B,H,HD,T])
    cgemm_kernel<3, false><<<g_d, blk, 0, stream>>>(nr, ni, s_wq_r, s_wq_i, nullptr, nullptr,
        nullptr, nullptr, qr, qi, cM, cD, cD);
    cgemm_kernel<3, false><<<g_d, blk, 0, stream>>>(nr, ni, s_wk_r, s_wk_i, nullptr, nullptr,
        nullptr, nullptr, kr, ki, cM, cD, cD);
    cgemm_kernel<4, false><<<g_d, blk, 0, stream>>>(nr, ni, s_wv_r, s_wv_i, nullptr, nullptr,
        nullptr, nullptr, vtr, vti, cM, cD, cD);

    attention_kernel<<<g_att, blk, 0, stream>>>(qr, qi, kr, ki, vtr, vti, aor, aoi);

    // output projection + bias + residual into xr/xi
    cgemm_kernel<1, false><<<g_d, blk, 0, stream>>>(aor, aoi, s_wo_r, s_wo_i,
        bo_r + l * cD, bo_i + l * cD, xr, xi, nullptr, nullptr, cM, cD, cD);

    // LN2
    c_layernorm_kernel<<<g_ln, blk, 0, stream>>>(xr, xi,
        ln2_gr + l * cD, ln2_gi + l * cD, ln2_br + l * cD, ln2_bi + l * cD, nr, ni);

    // FF1 (+bias, CReLU) -> bf16 hidden
    cgemm_kernel<0, true><<<g_ff, blk, 0, stream>>>(nr, ni, s_w1_r, s_w1_i,
        b1_r + l * cFF, b1_i + l * cFF, nullptr, nullptr, hr, hi, cM, cFF, cD);

    // FF2 (+bias) + residual into xr/xi
    cgemm_kernel<1, false><<<g_d, blk, 0, stream>>>(hr, hi, s_w2_r, s_w2_i,
        b2_r + l * cD, b2_i + l * cD, xr, xi, nullptr, nullptr, cM, cD, cFF);
  }

  // final LN + head (interleaved re/im fp32 to d_out, n<1001 guard)
  c_layernorm_kernel<<<g_ln, blk, 0, stream>>>(xr, xi, lnf_gr, lnf_gi, lnf_br, lnf_bi, nr, ni);
  cgemm_kernel<2, false><<<g_vo, blk, 0, stream>>>(nr, ni, s_wh_r, s_wh_i, bh_r, bh_i,
      (float*)d_out, nullptr, nullptr, nullptr, cM, cVO, cD);
}
